// SentenceHistoryAttn_60911226192257
// MI455X (gfx1250) — matrix operand order repaired and checked by reading, not compiled, not run on hardware
//
#include <hip/hip_runtime.h>

typedef __attribute__((ext_vector_type(16))) _Float16 v16h;
typedef __attribute__((ext_vector_type(8)))  _Float16 v8h;
typedef __attribute__((ext_vector_type(4)))  _Float16 v4h;
typedef __attribute__((ext_vector_type(8)))  float    v8f;
typedef __attribute__((ext_vector_type(4)))  float    v4f;

typedef union { v16h v; v8h h[2]; } frag16;

#define NTOK 8192
#define DDIM 1024
#define WPB  8   // waves per block (256 threads, wave32)

// ---------------------------------------------------------------------------
// CDNA5 plain (non-transpose) 128-bit global load with immediate offset: 16x16 f16 tile -> fragment.
// KOB is a byte offset folded into the VMEM offset field (24-bit signed).
// ---------------------------------------------------------------------------
template<int KOB>
__device__ __forceinline__ v8h load_tr16(const _Float16* p) {
    v8h t;
    asm volatile("global_load_b128 %0, %1, off offset:%2"
                 : "=v"(t) : "v"(p), "n"(KOB));
    return t;
}

__device__ __forceinline__ v8f wmma_f16(v16h a, v16h b, v8f c) {
    return __builtin_amdgcn_wmma_f32_16x16x32_f16(false, a, false, b,
                                                  (short)0, c, false, false);
}

// ---------------------------------------------------------------------------
// One K=32 step of the 32x64 wave tile. KOB = byte offset of this K-step
// relative to the current base pointers (constant -> VMEM immediates).
// 12 loads, 1 wait, 8 WMMAs.
// ---------------------------------------------------------------------------
template<int KOB>
__device__ __forceinline__ void kstep_32x64(const _Float16* ap0, const _Float16* ap1,
                                            const _Float16* bp0, const _Float16* bp1,
                                            const _Float16* bp2, const _Float16* bp3,
                                            v8f c[2][4]) {
    constexpr int KOE = KOB / 2;   // element offset
    frag16 a0, a1;
    a0.h[0] = *(const v8h*)(ap0 + KOE);
    a0.h[1] = *(const v8h*)(ap0 + KOE + 16);
    a1.h[0] = *(const v8h*)(ap1 + KOE);
    a1.h[1] = *(const v8h*)(ap1 + KOE + 16);

    frag16 b[4];
    b[0].h[0] = load_tr16<KOB     >(bp0);
    b[0].h[1] = load_tr16<KOB + 32>(bp0);
    b[1].h[0] = load_tr16<KOB     >(bp1);
    b[1].h[1] = load_tr16<KOB + 32>(bp1);
    b[2].h[0] = load_tr16<KOB     >(bp2);
    b[2].h[1] = load_tr16<KOB + 32>(bp2);
    b[3].h[0] = load_tr16<KOB     >(bp3);
    b[3].h[1] = load_tr16<KOB + 32>(bp3);

    // single wait for the 8 B loads; register ties keep the WMMAs below it
    asm volatile("s_wait_loadcnt 0"
                 : "+v"(b[0].h[0]), "+v"(b[0].h[1]), "+v"(b[1].h[0]), "+v"(b[1].h[1]),
                   "+v"(b[2].h[0]), "+v"(b[2].h[1]), "+v"(b[3].h[0]), "+v"(b[3].h[1]));

#pragma unroll
    for (int nt = 0; nt < 4; ++nt) {
        c[0][nt] = wmma_f16(a0.v, b[nt].v, c[0][nt]);
        c[1][nt] = wmma_f16(a1.v, b[nt].v, c[1][nt]);
    }
}

// ---------------------------------------------------------------------------
// Wave-level GEMM core: C[32x64] += A[m0..m0+31][:] * B[n0..n0+63][:]^T
// A, B row-major [*, DDIM] f16; both L2-resident on MI455X (192 MB L2).
// ---------------------------------------------------------------------------
__device__ __forceinline__ void wave_gemm_32x64(const _Float16* __restrict__ A,
                                                const _Float16* __restrict__ B,
                                                int m0, int n0, int lane,
                                                v8f c[2][4]) {
    const int ar = lane & 15;            // fragment row
    const int kh = (lane >> 4) * 8;      // K sub-offset per ISA A/B layout

    const _Float16* ap0 = A + (size_t)(m0 + ar) * DDIM + kh;
    const _Float16* ap1 = ap0 + (size_t)16 * DDIM;
    const _Float16* bp0 = B + (size_t)(n0 + ar) * DDIM + kh;
    const _Float16* bp1 = bp0 + (size_t)16 * DDIM;
    const _Float16* bp2 = bp0 + (size_t)32 * DDIM;
    const _Float16* bp3 = bp0 + (size_t)48 * DDIM;

    for (int kk = 0; kk < DDIM; kk += 128) {    // 8 iterations, 4 K-steps each
        kstep_32x64<  0>(ap0, ap1, bp0, bp1, bp2, bp3, c);
        kstep_32x64< 64>(ap0, ap1, bp0, bp1, bp2, bp3, c);
        kstep_32x64<128>(ap0, ap1, bp0, bp1, bp2, bp3, c);
        kstep_32x64<192>(ap0, ap1, bp0, bp1, bp2, bp3, c);
        ap0 += 128; ap1 += 128;
        bp0 += 128; bp1 += 128; bp2 += 128; bp3 += 128;
    }
}

// ---------------------------------------------------------------------------
// f32 -> f16 conversion, 4 elements/thread
// ---------------------------------------------------------------------------
__global__ __launch_bounds__(256) void cvt_f16_kernel(const float* __restrict__ src,
                                                      _Float16* __restrict__ dst,
                                                      int n4) {
    int i = blockIdx.x * 256 + threadIdx.x;
    if (i < n4) {
        v4f s = ((const v4f*)src)[i];
        v4h d;
#pragma unroll
        for (int k = 0; k < 4; ++k) d[k] = (_Float16)s[k];
        ((v4h*)dst)[i] = d;
    }
}

// ---------------------------------------------------------------------------
// Kernel 1: x_proj = relu(x @ W^T + b); one wave -> 32x64 f16 output block
// ---------------------------------------------------------------------------
__global__ __launch_bounds__(256) void proj_kernel(const _Float16* __restrict__ xh,
                                                   const _Float16* __restrict__ wh,
                                                   const float* __restrict__ bias,
                                                   _Float16* __restrict__ xph) {
    const int lane = threadIdx.x & 31;
    const int gw   = blockIdx.x * WPB + (threadIdx.x >> 5);
    const int wavesPerN = DDIM / 64;               // 16
    const int m0 = (gw / wavesPerN) * 32;
    const int n0 = (gw % wavesPerN) * 64;

    v8f c[2][4] = {};
    wave_gemm_32x64(xh, wh, m0, n0, lane, c);

    const int mloc = (lane >> 4) << 3;             // C/D layout: M = v + (lane/16)*8
    const int nloc = lane & 15;
#pragma unroll
    for (int nt = 0; nt < 4; ++nt) {
        const int col = n0 + nt * 16 + nloc;
        const float bv = bias[col];
#pragma unroll
        for (int mt = 0; mt < 2; ++mt) {
            _Float16* op = xph + (size_t)(m0 + mt * 16 + mloc) * DDIM + col;
#pragma unroll
            for (int v = 0; v < 8; ++v) {
                float val = fmaxf(c[mt][nt][v] + bv, 0.0f);
                op[(size_t)v * DDIM] = (_Float16)val;
            }
        }
    }
}

// ---------------------------------------------------------------------------
// Kernel 2: scores = x_proj @ x_proj^T + causal mask + recency (f32 out).
// Wave blocks strictly above the diagonal skip the GEMM and write -inf.
// ---------------------------------------------------------------------------
__global__ __launch_bounds__(256) void scores_kernel(const _Float16* __restrict__ xph,
                                                     const float* __restrict__ recw,
                                                     float* __restrict__ out) {
    const int lane = threadIdx.x & 31;
    const int gw   = blockIdx.x * WPB + (threadIdx.x >> 5);
    const int wavesPerN = NTOK / 64;               // 128
    const int m0 = (gw / wavesPerN) * 32;
    const int n0 = (gw % wavesPerN) * 64;

    const int mloc = (lane >> 4) << 3;
    const int nloc = lane & 15;

    if (n0 > m0 + 31) {                            // fully masked 32x64 block
        const float ninf = -__builtin_inff();
#pragma unroll
        for (int mt = 0; mt < 2; ++mt) {
            float* op = out + (size_t)(m0 + mt * 16 + mloc) * NTOK + n0 + nloc;
#pragma unroll
            for (int v = 0; v < 8; ++v)
#pragma unroll
                for (int nt = 0; nt < 4; ++nt)
                    op[(size_t)v * NTOK + nt * 16] = ninf;
        }
        return;
    }

    v8f c[2][4] = {};
    wave_gemm_32x64(xph, xph, m0, n0, lane, c);

    const float rw = recw[0];
#pragma unroll
    for (int mt = 0; mt < 2; ++mt) {
        const int ib = m0 + mt * 16 + mloc;
        float* op = out + (size_t)ib * NTOK + n0 + nloc;
#pragma unroll
        for (int v = 0; v < 8; ++v) {
            const int i = ib + v;
#pragma unroll
            for (int nt = 0; nt < 4; ++nt) {
                const int j = n0 + nt * 16 + nloc;
                float s = (j > i) ? -__builtin_inff()
                                  : (c[mt][nt][v] + rw * (float)(i - j));
                op[(size_t)v * NTOK + nt * 16] = s;
            }
        }
    }
}

// ---------------------------------------------------------------------------
// Kernel 3: in-place row softmax; one block per row, 32 elems/thread in
// registers -> one read + one write of the 256 MB matrix.
// ---------------------------------------------------------------------------
__global__ __launch_bounds__(256) void softmax_kernel(float* __restrict__ out) {
    __shared__ float red[256];
    const int t = threadIdx.x;
    float* rp = out + (size_t)blockIdx.x * NTOK;

    float v[32];
    float m = -__builtin_inff();
#pragma unroll
    for (int k = 0; k < 32; ++k) {
        v[k] = rp[k * 256 + t];
        m = fmaxf(m, v[k]);
    }
    red[t] = m;
    __syncthreads();
    for (int s = 128; s > 0; s >>= 1) {
        if (t < s) red[t] = fmaxf(red[t], red[t + s]);
        __syncthreads();
    }
    const float rowmax = red[0];
    __syncthreads();

    float sum = 0.0f;
#pragma unroll
    for (int k = 0; k < 32; ++k) {
        v[k] = __expf(v[k] - rowmax);   // exp(-inf - max) == 0 for masked cols
        sum += v[k];
    }
    red[t] = sum;
    __syncthreads();
    for (int s = 128; s > 0; s >>= 1) {
        if (t < s) red[t] += red[t + s];
        __syncthreads();
    }
    const float inv = 1.0f / red[0];
#pragma unroll
    for (int k = 0; k < 32; ++k) rp[k * 256 + t] = v[k] * inv;
}

// ---------------------------------------------------------------------------
extern "C" void kernel_launch(void* const* d_in, const int* in_sizes, int n_in,
                              void* d_out, int out_size, void* d_ws, size_t ws_size,
                              hipStream_t stream) {
    const float* x    = (const float*)d_in[0];   // [NTOK, DDIM]
    const float* W    = (const float*)d_in[1];   // [DDIM, DDIM]
    const float* bias = (const float*)d_in[2];   // [DDIM]
    const float* recw = (const float*)d_in[3];   // [1]
    float* out = (float*)d_out;                  // [NTOK, NTOK]

    char* ws = (char*)d_ws;
    _Float16* xh  = (_Float16*)ws;                                           // 16 MB
    _Float16* wh  = (_Float16*)(ws + (size_t)NTOK * DDIM * 2);               //  2 MB
    _Float16* xph = (_Float16*)(ws + (size_t)NTOK * DDIM * 2
                                   + (size_t)DDIM * DDIM * 2);               // 16 MB

    const int nx4 = NTOK * DDIM / 4;
    const int nw4 = DDIM * DDIM / 4;
    cvt_f16_kernel<<<(nx4 + 255) / 256, 256, 0, stream>>>(x, xh, nx4);
    cvt_f16_kernel<<<(nw4 + 255) / 256, 256, 0, stream>>>(W, wh, nw4);

    const int projWaves  = (NTOK / 32) * (DDIM / 64);   // 4096
    const int scoreWaves = (NTOK / 32) * (NTOK / 64);   // 32768
    proj_kernel  <<<projWaves  / WPB, 256, 0, stream>>>(xh, wh, bias, xph);
    scores_kernel<<<scoreWaves / WPB, 256, 0, stream>>>(xph, recw, out);
    softmax_kernel<<<NTOK, 256, 0, stream>>>(out);
}
